// GDM_38508676776442
// MI455X (gfx1250) — compile-verified, matrix operand order checked by reading, not run on hardware
//
#include <hip/hip_runtime.h>
#include <hip/hip_bf16.h>
#include <math.h>

typedef _Float16 v8h  __attribute__((ext_vector_type(8)));
typedef _Float16 v16h __attribute__((ext_vector_type(16)));
typedef float    v8f  __attribute__((ext_vector_type(8)));

#define BLK_M 128
#define BLK_N 256
#define BLK_K 32
#define LDSP  40   // 32 + 8 halves pad; 80B row stride keeps 16B alignment

// Async copy of 16 bytes global -> LDS (gfx1250, ASYNCcnt-tracked).
__device__ __forceinline__ void async_b128(void* lds, const void* gptr) {
  const unsigned lds_off = (unsigned)(unsigned long long)lds;      // flat LDS addr low bits = LDS offset
  const unsigned long long ga = (unsigned long long)gptr;
  asm volatile("global_load_async_to_lds_b128 %0, %1, off"
               :: "v"(lds_off), "v"(ga) : "memory");
}
__device__ __forceinline__ void wait_async0() {
  asm volatile("s_wait_asynccnt 0x0" ::: "memory");
}

struct GemmP {
  const _Float16* A; const _Float16* B; void* C;
  long long sAo, sAi, sBo, sBi, sCo, sCi;   // batch strides (z = zo*zdiv + zi)
  int lda, ldb, ldc;
  int M, N, K;
  int zdiv;
  int accum;    // add pre-existing f32 C
  int gelu;     // exact gelu epilogue
  int outF16;   // store _Float16 else float
  int transC;   // store C[n*ldc + m] instead of C[m*ldc + n]
  float scale; const float* scale_ptr;
};

// ---------------- generic batched WMMA GEMM (B always k-contiguous: B[n*ldb+k]) ----------------
__global__ __launch_bounds__(256) void gemm_wmma_f16(GemmP p) {
  __shared__ _Float16 As[BLK_M][LDSP];
  __shared__ _Float16 Bs[BLK_N][LDSP];

  const int tid  = threadIdx.x;
  const int lane = tid & 31;
  const int wave = tid >> 5;
  const int wm = (wave >> 2) * 64;   // 0,64
  const int wn = (wave & 3) * 64;    // 0,64,128,192

  const int z  = blockIdx.z;
  const int zo = z / p.zdiv, zi = z % p.zdiv;
  const _Float16* A = p.A + (long long)zo * p.sAo + (long long)zi * p.sAi;
  const _Float16* B = p.B + (long long)zo * p.sBo + (long long)zi * p.sBi;
  const long long coff = (long long)zo * p.sCo + (long long)zi * p.sCi;

  const int m0 = blockIdx.y * BLK_M;
  const int n0 = blockIdx.x * BLK_N;

  v8f acc[4][4] = {};
  const float scl = p.scale * (p.scale_ptr ? p.scale_ptr[0] : 1.0f);

  for (int k0 = 0; k0 < p.K; k0 += BLK_K) {
    // ---- stage A tile (row-major M x K): 2 threads/row, 2 async b128 each ----
    {
      const int r  = tid >> 1;
      const int kh = (tid & 1) * 16;
      const int gm = m0 + r;
      if (gm < p.M) {
        const _Float16* src = A + (long long)gm * p.lda + k0 + kh;
        async_b128(&As[r][kh],     src);
        async_b128(&As[r][kh + 8], src + 8);
      } else {
        *(v8h*)&As[r][kh]     = (v8h){};
        *(v8h*)&As[r][kh + 8] = (v8h){};
      }
    }
    // ---- stage B tile (k-contiguous: B[n*ldb+k]): 1 thread/row, 4 async b128 each ----
    {
      const int nl = tid;              // 0..255
      const int gn = n0 + nl;
      if (gn < p.N) {
        const _Float16* src = B + (long long)gn * p.ldb + k0;
        async_b128(&Bs[nl][0],  src);
        async_b128(&Bs[nl][8],  src + 8);
        async_b128(&Bs[nl][16], src + 16);
        async_b128(&Bs[nl][24], src + 24);
      } else {
        *(v8h*)&Bs[nl][0]  = (v8h){};
        *(v8h*)&Bs[nl][8]  = (v8h){};
        *(v8h*)&Bs[nl][16] = (v8h){};
        *(v8h*)&Bs[nl][24] = (v8h){};
      }
    }
    wait_async0();
    __syncthreads();

    // ---- fragments per ISA 16-bit layout ----
    const int lrow = lane & 15;
    const int koff = (lane & 16) ? 8 : 0;
    v16h afr[4];
    #pragma unroll
    for (int tm = 0; tm < 4; ++tm) {
      v8h lo = *(const v8h*)&As[wm + tm * 16 + lrow][koff];
      v8h hi = *(const v8h*)&As[wm + tm * 16 + lrow][koff + 16];
      afr[tm] = __builtin_shufflevector(lo, hi, 0,1,2,3,4,5,6,7,8,9,10,11,12,13,14,15);
    }
    #pragma unroll
    for (int tn = 0; tn < 4; ++tn) {
      v8h lo = *(const v8h*)&Bs[wn + tn * 16 + lrow][koff];
      v8h hi = *(const v8h*)&Bs[wn + tn * 16 + lrow][koff + 16];
      v16h bfr = __builtin_shufflevector(lo, hi, 0,1,2,3,4,5,6,7,8,9,10,11,12,13,14,15);
      #pragma unroll
      for (int tm = 0; tm < 4; ++tm)
        acc[tm][tn] = __builtin_amdgcn_wmma_f32_16x16x32_f16(
            false, afr[tm], false, bfr, (short)0, acc[tm][tn], false, false);
    }
    __syncthreads();
  }

  // ---- epilogue: lane(0..15)->N, VGPR r -> M=r (+8 for hi lanes) ----
  const int lrow  = lane & 15;
  const int rbase = (lane & 16) ? 8 : 0;
  #pragma unroll
  for (int tm = 0; tm < 4; ++tm) {
    #pragma unroll
    for (int tn = 0; tn < 4; ++tn) {
      #pragma unroll
      for (int r = 0; r < 8; ++r) {
        const int m = m0 + wm + tm * 16 + rbase + r;
        const int n = n0 + wn + tn * 16 + lrow;
        if (m < p.M && n < p.N) {
          float v = acc[tm][tn][r] * scl;
          const long long ci = p.transC ? coff + (long long)n * p.ldc + m
                                        : coff + (long long)m * p.ldc + n;
          if (p.accum) v += ((float*)p.C)[ci];
          if (p.gelu)  v = 0.5f * v * (1.0f + erff(v * 0.70710678118654752f));
          if (p.outF16) ((_Float16*)p.C)[ci] = (_Float16)v;
          else          ((float*)p.C)[ci]    = v;
        }
      }
    }
  }
}

// ---------------- helpers ----------------
__global__ __launch_bounds__(256) void cvt_f32_f16(const float* __restrict__ s,
                                                   _Float16* __restrict__ d, long long n) {
  long long i = (long long)blockIdx.x * blockDim.x + threadIdx.x;
  const long long st = (long long)gridDim.x * blockDim.x;
  for (; i < n; i += st) d[i] = (_Float16)s[i];
}

// d[h][c][r] = s[h][r][c]  (per-head transpose, R x C -> C x R), f32 -> f16
__global__ __launch_bounds__(256) void cvt_t_f32_f16(const float* __restrict__ s,
                                                     _Float16* __restrict__ d,
                                                     int R, int C, long long total) {
  long long i = (long long)blockIdx.x * blockDim.x + threadIdx.x;
  const long long st = (long long)gridDim.x * blockDim.x;
  const long long mat = (long long)R * C;
  for (; i < total; i += st) {
    const long long h = i / mat;
    const long long rem = i - h * mat;
    const int c = (int)(rem / R);
    const int r = (int)(rem - (long long)c * R);
    d[i] = (_Float16)s[h * mat + (long long)r * C + c];
  }
}

// mcol[e] = sum_v W_e[v,e] / (V + 1e-8)
__global__ __launch_bounds__(256) void colmean_k(const float* __restrict__ We,
                                                 float* __restrict__ mcol, int V, int E) {
  const int e = blockIdx.x * blockDim.x + threadIdx.x;
  if (e >= E) return;
  float s = 0.0f;
  for (int v = 0; v < V; ++v) s += We[(long long)v * E + e];
  mcol[e] = s / ((float)V + 1e-8f);
}

// v_h[b,s,e] = W_e[x[b,s],e] - mcol[e]
__global__ __launch_bounds__(256) void embed_k(const int* __restrict__ x,
                                               const float* __restrict__ We,
                                               const float* __restrict__ mcol,
                                               _Float16* __restrict__ vh,
                                               long long total, int E) {
  long long i = (long long)blockIdx.x * blockDim.x + threadIdx.x;
  const long long st = (long long)gridDim.x * blockDim.x;
  for (; i < total; i += st) {
    const int e = (int)(i % E);
    const long long t = i / E;
    const long long tok = (long long)x[t];
    vh[i] = (_Float16)(We[tok * E + e] - mcol[e]);
  }
}

// softmax over rows; grid = (rows, heads); P gets f16 probs
__global__ __launch_bounds__(256) void softmax_k(const float* __restrict__ S,
                                                 _Float16* __restrict__ P, int cols) {
  const long long base = ((long long)blockIdx.y * gridDim.x + blockIdx.x) * cols;
  __shared__ float red[256];
  const int tid = threadIdx.x;
  float mx = -3.4e38f;
  for (int c = tid; c < cols; c += 256) mx = fmaxf(mx, S[base + c]);
  red[tid] = mx; __syncthreads();
  for (int s = 128; s > 0; s >>= 1) { if (tid < s) red[tid] = fmaxf(red[tid], red[tid + s]); __syncthreads(); }
  const float m = red[0]; __syncthreads();
  float sum = 0.0f;
  for (int c = tid; c < cols; c += 256) sum += __expf(S[base + c] - m);
  red[tid] = sum; __syncthreads();
  for (int s = 128; s > 0; s >>= 1) { if (tid < s) red[tid] += red[tid + s]; __syncthreads(); }
  const float inv = 1.0f / red[0];
  for (int c = tid; c < cols; c += 256) P[base + c] = (_Float16)(__expf(S[base + c] - m) * inv);
}

// dB[b,e] = B_lr * sum_s v[b,s,e]
__global__ __launch_bounds__(256) void db_k(const _Float16* __restrict__ vh,
                                            const float* __restrict__ Blr,
                                            float* __restrict__ dB, int Bn, int S, int E) {
  const int i = blockIdx.x * blockDim.x + threadIdx.x;
  if (i >= Bn * E) return;
  const int b = i / E, e = i % E;
  float s = 0.0f;
  for (int t = 0; t < S; ++t) s += (float)vh[((long long)b * S + t) * E + e];
  dB[i] = s * Blr[0];
}

// fin[b,q,e] = (dA[b,q,e] + dB[b,e]) / S  (f16)
__global__ __launch_bounds__(256) void fin_k(const float* __restrict__ dA,
                                             const float* __restrict__ dB,
                                             _Float16* __restrict__ fin,
                                             long long total, int Qr, int E, float invS) {
  long long i = (long long)blockIdx.x * blockDim.x + threadIdx.x;
  const long long st = (long long)gridDim.x * blockDim.x;
  for (; i < total; i += st) {
    const int e = (int)(i % E);
    const long long bq = i / E;
    const int b = (int)(bq / Qr);
    fin[i] = (_Float16)((dA[i] + dB[b * E + e]) * invS);
  }
}

// ---------------- host orchestration ----------------
extern "C" void kernel_launch(void* const* d_in, const int* in_sizes, int n_in,
                              void* d_out, int out_size, void* d_ws, size_t ws_size,
                              hipStream_t stream) {
  (void)in_sizes; (void)n_in; (void)out_size; (void)ws_size;
  constexpr int Bn = 4, S = 1024, E = 768, H = 12, V = 50257, DFF = 3072;
  constexpr int Qr = S + 1; // 1025

  const int*   x    = (const int*)  d_in[0];
  const float* W_e  = (const float*)d_in[1];
  const float* W_p  = (const float*)d_in[2];
  const float* W_q  = (const float*)d_in[3];
  const float* W_k  = (const float*)d_in[4];
  const float* W_v  = (const float*)d_in[5];
  const float* A_lr = (const float*)d_in[6];
  const float* B_lr = (const float*)d_in[7];
  const float* ffw1 = (const float*)d_in[8];
  const float* ffw2 = (const float*)d_in[9];
  float* out = (float*)d_out;

  char* ws = (char*)d_ws; size_t off = 0;
  auto carve = [&](size_t bytes) { void* p = ws + off; off += (bytes + 255) & ~size_t(255); return p; };

  _Float16* We_h   = (_Float16*)carve((size_t)V * E * 2);
  _Float16* p_h    = (_Float16*)carve((size_t)Qr * E * 2);
  _Float16* WqT_h  = (_Float16*)carve((size_t)H * E * E * 2);   // [h][f][e]
  _Float16* WkT_h  = (_Float16*)carve((size_t)H * E * E * 2);
  _Float16* WvT_h  = (_Float16*)carve((size_t)H * E * E * 2);
  _Float16* w1_h   = (_Float16*)carve((size_t)DFF * E * 2);     // [j][e] already k-contig
  _Float16* w2_h   = (_Float16*)carve((size_t)DFF * E * 2);     // [e][j] already k-contig
  _Float16* v_h    = (_Float16*)carve((size_t)Bn * S * E * 2);
  _Float16* Q_h    = (_Float16*)carve((size_t)H * Qr * E * 2);
  _Float16* K_h    = (_Float16*)carve((size_t)H * S * E * 2);   // [h][kpos][e] k-contig for scores
  float*    sc_f   = (float*)   carve((size_t)H * Qr * S * 4);
  _Float16* at_h   = (_Float16*)carve((size_t)H * Qr * S * 2);
  _Float16* VtT_h  = (_Float16*)carve((size_t)Bn * H * E * S * 2); // [b][h][e][s] (transC store)
  float*    dA_f   = (float*)   carve((size_t)Bn * Qr * E * 4);
  float*    dB_f   = (float*)   carve((size_t)Bn * E * 4);
  _Float16* fin_h  = (_Float16*)carve((size_t)Bn * Qr * E * 2);
  _Float16* g_h    = (_Float16*)carve((size_t)Bn * Qr * DFF * 2);
  _Float16* f2_h   = (_Float16*)carve((size_t)Bn * Qr * E * 2);
  float*    mcol   = (float*)   carve((size_t)E * 4);

  cvt_f32_f16<<<2048, 256, 0, stream>>>(W_e,  We_h, (long long)V * E);
  cvt_f32_f16<<<2048, 256, 0, stream>>>(W_p,  p_h,  (long long)Qr * E);
  cvt_f32_f16<<<2048, 256, 0, stream>>>(ffw1, w1_h, (long long)DFF * E);
  cvt_f32_f16<<<2048, 256, 0, stream>>>(ffw2, w2_h, (long long)DFF * E);
  cvt_t_f32_f16<<<2048, 256, 0, stream>>>(W_q, WqT_h, E, E, (long long)H * E * E);
  cvt_t_f32_f16<<<2048, 256, 0, stream>>>(W_k, WkT_h, E, E, (long long)H * E * E);
  cvt_t_f32_f16<<<2048, 256, 0, stream>>>(W_v, WvT_h, E, E, (long long)H * E * E);

  colmean_k<<<(E + 255) / 256, 256, 0, stream>>>(W_e, mcol, V, E);
  embed_k<<<2048, 256, 0, stream>>>(x, W_e, mcol, v_h, (long long)Bn * S * E, E);

  auto gemm = [&](const _Float16* A, long long sAo, long long sAi, int lda,
                  const _Float16* B, long long sBo, long long sBi, int ldb,
                  void* C, long long sCo, long long sCi, int ldc,
                  int M, int N, int K, int nz, int zdiv,
                  int accum, int gelu, int outF16, int transC,
                  float scale, const float* scale_ptr) {
    GemmP p{A, B, C, sAo, sAi, sBo, sBi, sCo, sCi, lda, ldb, ldc,
            M, N, K, zdiv, accum, gelu, outF16, transC, scale, scale_ptr};
    dim3 grid((N + BLK_N - 1) / BLK_N, (M + BLK_M - 1) / BLK_M, nz);
    gemm_wmma_f16<<<grid, 256, 0, stream>>>(p);
  };

  // Q[h] = p @ W_q[h] : B = WqT[h] (n=f, k=e contiguous)
  gemm(p_h, 0, 0, E,  WqT_h, (long long)E * E, 0, E,
       Q_h, (long long)Qr * E, 0, E,  Qr, E, E, H, 1, 0, 0, 1, 0, 1.0f, nullptr);
  // K[h] = p[:-1] @ W_k[h]
  gemm(p_h, 0, 0, E,  WkT_h, (long long)E * E, 0, E,
       K_h, (long long)S * E, 0, E,  S, E, E, H, 1, 0, 0, 1, 0, 1.0f, nullptr);
  // scores[h] = Q[h] @ K[h]^T / sqrt(E) : B = K_h[h] (n=kpos, k=e contiguous)
  gemm(Q_h, (long long)Qr * E, 0, E,  K_h, (long long)S * E, 0, E,
       sc_f, (long long)Qr * S, 0, S,  Qr, S, E, H, 1, 0, 0, 0, 0,
       1.0f / sqrtf((float)E), nullptr);
  softmax_k<<<dim3(Qr, H), 256, 0, stream>>>(sc_f, at_h, S);
  // Vt[b,h] = v[b] @ W_v[h], stored transposed: VtT[b,h][e][s]  (z = b*H + h)
  gemm(v_h, (long long)S * E, 0, E,  WvT_h, 0, (long long)E * E, E,
       VtT_h, (long long)H * E * S, (long long)E * S, S,
       S, E, E, Bn * H, H, 0, 0, 1, 1, 1.0f, nullptr);
  // dA[b] = sum_h A_lr[h] * (attn[h] @ Vt[b,h]) : B = VtT[b,h] (n=e, k=s contiguous)
  hipMemsetAsync(dA_f, 0, (size_t)Bn * Qr * E * 4, stream);
  for (int h = 0; h < H; ++h) {
    gemm(at_h + (long long)h * Qr * S, 0, 0, S,
         VtT_h + (long long)h * E * S, (long long)H * E * S, 0, S,
         dA_f, (long long)Qr * E, 0, E,
         Qr, E, S, Bn, 1, 1, 0, 0, 0, 1.0f, A_lr + h);
  }
  db_k<<<(Bn * E + 255) / 256, 256, 0, stream>>>(v_h, B_lr, dB_f, Bn, S, E);
  fin_k<<<2048, 256, 0, stream>>>(dA_f, dB_f, fin_h, (long long)Bn * Qr * E, Qr, E, 1.0f / (float)S);
  // FF1: g = gelu(fin @ ff_w1^T) : B = w1_h (n=j, k=e contiguous)
  gemm(fin_h, 0, 0, E,  w1_h, 0, 0, E,
       g_h, 0, 0, DFF,  Bn * Qr, DFF, E, 1, 1, 0, 1, 1, 0, 1.0f, nullptr);
  // FF2: f2 = g @ ff_w2^T : B = w2_h (n=e, k=j contiguous)
  gemm(g_h, 0, 0, DFF,  w2_h, 0, 0, DFF,
       f2_h, 0, 0, E,  Bn * Qr, E, DFF, 1, 1, 0, 0, 1, 0, 1.0f, nullptr);
  // logits[b] = f2[b,:-1] @ W_e^T : B = We_h (n=v, k=e contiguous); M=1024 skips last row
  gemm(f2_h, (long long)Qr * E, 0, E,  We_h, 0, 0, E,
       out, (long long)S * V, 0, V,  S, V, E, Bn, 1, 0, 0, 0, 0, 1.0f, nullptr);
}
